// Head_38199439131113
// MI455X (gfx1250) — matrix operand order repaired and checked by reading, not compiled, not run on hardware
//
#include <hip/hip_runtime.h>

typedef __bf16 bf16_t;
typedef __attribute__((ext_vector_type(16))) __bf16 v16bf;
typedef __attribute__((ext_vector_type(8)))  __bf16 v8bf;
typedef __attribute__((ext_vector_type(8)))  float  v8f;

#define NB 8
#define T 2048
#define C 1024
#define D 128
#define SCALE 0.08838834764831845f   // 1/sqrt(128)

// ---------------------------------------------------------------------------
// Kernel 0: f32 -> bf16 conversion (x and the three weight matrices), done
// ONCE so the GEMM inner loops are pure bf16 load + WMMA.
// ---------------------------------------------------------------------------
__global__ __launch_bounds__(256)
void cvt_f32_to_bf16_kernel(const float* __restrict__ src,
                            bf16_t* __restrict__ dst, int n)
{
    const int i = (blockIdx.x * 256 + threadIdx.x) * 8;
    if (i >= n) return;
    const float4 a = *(const float4*)(src + i);
    const float4 b = *(const float4*)(src + i + 4);
    v8bf o;
    o[0] = (bf16_t)a.x; o[1] = (bf16_t)a.y; o[2] = (bf16_t)a.z; o[3] = (bf16_t)a.w;
    o[4] = (bf16_t)b.x; o[5] = (bf16_t)b.y; o[6] = (bf16_t)b.z; o[7] = (bf16_t)b.w;
    *(v8bf*)(dst + i) = o;
}

// ---------------------------------------------------------------------------
// Kernel 1: QKV projection, all-bf16.  Y = x @ W^T.
// One wave = one 16-row tile of one matrix.  Explicit double buffering with
// two disjoint fragment sets (A/B) so next-step loads overlap the current
// WMMA burst; __launch_bounds__(...,1) lets the backend keep both sets live.
// ---------------------------------------------------------------------------
__global__ __launch_bounds__(256, 1)
void proj_qkv_kernel(const bf16_t* __restrict__ xb,
                     const bf16_t* __restrict__ Wqb,
                     const bf16_t* __restrict__ Wkb,
                     const bf16_t* __restrict__ Wvb,
                     bf16_t* __restrict__ Qb,
                     bf16_t* __restrict__ Kb,
                     bf16_t* __restrict__ Vt)
{
    const int lane = threadIdx.x & 31;
    const int half = lane >> 4;
    const int l16  = lane & 15;
    const int gid  = blockIdx.x * 8 + (threadIdx.x >> 5);   // 3072 wave tasks
    const int matrix   = gid >> 10;                          // 0=Q 1=K 2=V
    const int row_base = (gid & 1023) * 16;

    const bf16_t* __restrict__ W =
        (matrix == 0) ? Wqb : (matrix == 1) ? Wkb : Wvb;
    const bf16_t* __restrict__ wbase = W + (size_t)l16 * C + half * 16;
    const bf16_t* __restrict__ xrow  = xb + (size_t)(row_base + l16) * C;

    auto load_set = [&](int kc, v16bf& a, v16bf* bf) {
        const v8bf lo = *(const v8bf*)(xrow + kc + half * 16);
        const v8bf hi = *(const v8bf*)(xrow + kc + half * 16 + 8);
        #pragma unroll
        for (int ot = 0; ot < 8; ++ot)
            bf[ot] = *(const v16bf*)(wbase + (size_t)(ot * 16) * C + kc);
        a = __builtin_shufflevector(lo, hi,
            0,1,2,3,4,5,6,7,8,9,10,11,12,13,14,15);
    };

    v8f acc[8] = {};
    v16bf aA, aB, bA[8], bB[8];

    load_set(0, aA, bA);
    #pragma unroll 1
    for (int kc = 0; kc < C; kc += 64) {
        load_set(kc + 32, aB, bB);              // prefetch set B
        #pragma unroll
        for (int ot = 0; ot < 8; ++ot)          // compute on set A
            acc[ot] = __builtin_amdgcn_wmma_f32_16x16x32_bf16(
                false, aA, false, bA[ot], (short)0, acc[ot], false, false);
        if (kc + 64 < C)
            load_set(kc + 64, aA, bA);          // prefetch next set A
        #pragma unroll
        for (int ot = 0; ot < 8; ++ot)          // compute on set B
            acc[ot] = __builtin_amdgcn_wmma_f32_16x16x32_bf16(
                false, aB, false, bB[ot], (short)0, acc[ot], false, false);
    }

    // C/D layout: VGPR r -> row m = r + 8*half, col n = l16.
    if (matrix == 0) {                  // Q, pre-scaled by 1/sqrt(d_k)
        #pragma unroll
        for (int ot = 0; ot < 8; ++ot)
            #pragma unroll
            for (int r = 0; r < 8; ++r)
                Qb[(size_t)(row_base + r + half * 8) * D + ot * 16 + l16] =
                    (bf16_t)(acc[ot][r] * SCALE);
    } else if (matrix == 1) {           // K
        #pragma unroll
        for (int ot = 0; ot < 8; ++ot)
            #pragma unroll
            for (int r = 0; r < 8; ++r)
                Kb[(size_t)(row_base + r + half * 8) * D + ot * 16 + l16] =
                    (bf16_t)acc[ot][r];
    } else {                            // V transposed [B,128,T]
        const int b  = row_base >> 11;
        const int t0 = row_base & (T - 1);
        #pragma unroll
        for (int ot = 0; ot < 8; ++ot)
            #pragma unroll
            for (int r = 0; r < 8; ++r)
                Vt[((size_t)b * D + ot * 16 + l16) * T + t0 + r + half * 8] =
                    (bf16_t)acc[ot][r];
    }
}

// ---------------------------------------------------------------------------
// Kernel 2: causal flash attention.  One wave owns a 16-query tile, streams
// kv in blocks of 32, online softmax, P routed via per-wave LDS to reshape
// C-layout -> A-layout for the P*V WMMA.  V fragments are issued BEFORE the
// softmax VALU block so their latency hides under it.
// ---------------------------------------------------------------------------
__global__ __launch_bounds__(256, 1)
void flash_attn_kernel(const bf16_t* __restrict__ Qb,
                       const bf16_t* __restrict__ Kb,
                       const bf16_t* __restrict__ Vt,
                       float* __restrict__ out)
{
    __shared__ bf16_t pbuf[8][16 * 32];  // 1 KB per-wave P staging

    const int wave = threadIdx.x >> 5;
    const int lane = threadIdx.x & 31;
    const int half = lane >> 4;
    const int l16  = lane & 15;

    const int task = blockIdx.x * 8 + wave;     // 1024 q tiles
    const int b  = task >> 7;
    const int q0 = (task & 127) * 16;

    // Q A-fragments (Q already carries the 1/sqrt(d) scale).
    v16bf qa[4];
    const bf16_t* __restrict__ qrow = Qb + ((size_t)b * T + q0 + l16) * D;
    #pragma unroll
    for (int c = 0; c < 4; ++c) {
        const v8bf lo = *(const v8bf*)(qrow + c * 32 + half * 16);
        const v8bf hi = *(const v8bf*)(qrow + c * 32 + half * 16 + 8);
        qa[c] = __builtin_shufflevector(lo, hi,
            0,1,2,3,4,5,6,7,8,9,10,11,12,13,14,15);
    }

    const bf16_t* __restrict__ kbase = Kb + ((size_t)b * T + l16) * D + half * 16;
    const bf16_t* __restrict__ vbase = Vt + ((size_t)b * D + l16) * T + half * 16;

    v8f   o_acc[8] = {};
    float m_st[8], l_st[8];
    #pragma unroll
    for (int r = 0; r < 8; ++r) { m_st[r] = -__builtin_inff(); l_st[r] = 0.f; }

    const int jmax = (q0 + 16 + 31) >> 5;

    #pragma unroll 1
    for (int jb = 0; jb < jmax; ++jb) {
        const int kv0 = jb * 32;
        const bool diag = (kv0 + 31 > q0);      // wave-uniform mask predicate

        // ---- stage all 8 K fragments, then burst 8 WMMAs (S = Q K^T) ----
        v16bf kf[8];
        #pragma unroll
        for (int nt = 0; nt < 2; ++nt)
            #pragma unroll
            for (int c = 0; c < 4; ++c)
                kf[nt * 4 + c] = *(const v16bf*)
                    (kbase + (size_t)(kv0 + nt * 16) * D + c * 32);
        v8f s[2] = {v8f{}, v8f{}};
        #pragma unroll
        for (int nt = 0; nt < 2; ++nt)
            #pragma unroll
            for (int c = 0; c < 4; ++c)
                s[nt] = __builtin_amdgcn_wmma_f32_16x16x32_bf16(
                    false, qa[c], false, kf[nt * 4 + c], (short)0, s[nt],
                    false, false);

        // ---- issue V fragment loads early: latency hides under softmax ----
        v16bf vf[8];
        #pragma unroll
        for (int vt = 0; vt < 8; ++vt)
            vf[vt] = *(const v16bf*)(vbase + (size_t)(vt * 16) * T + kv0);

        // ---- causal mask (diagonal blocks only) + online softmax ----
        #pragma unroll
        for (int r = 0; r < 8; ++r) {
            float v0 = s[0][r];
            float v1 = s[1][r];
            if (diag) {
                const int qg = q0 + r + half * 8;
                if (kv0 + l16      > qg) v0 = -__builtin_inff();
                if (kv0 + 16 + l16 > qg) v1 = -__builtin_inff();
            }
            float vm = fmaxf(v0, v1);
            vm = fmaxf(vm, __shfl_xor(vm, 1, 32));
            vm = fmaxf(vm, __shfl_xor(vm, 2, 32));
            vm = fmaxf(vm, __shfl_xor(vm, 4, 32));
            vm = fmaxf(vm, __shfl_xor(vm, 8, 32));
            const float mnew  = fmaxf(m_st[r], vm);
            const float alpha = __expf(m_st[r] - mnew);
            const float p0 = __expf(v0 - mnew);
            const float p1 = __expf(v1 - mnew);

            float ps = p0 + p1;
            ps += __shfl_xor(ps, 1, 32);
            ps += __shfl_xor(ps, 2, 32);
            ps += __shfl_xor(ps, 4, 32);
            ps += __shfl_xor(ps, 8, 32);

            l_st[r] = l_st[r] * alpha + ps;
            m_st[r] = mnew;
            #pragma unroll
            for (int vt = 0; vt < 8; ++vt) o_acc[vt][r] *= alpha;

            const int m = r + half * 8;
            pbuf[wave][m * 32 + l16]      = (bf16_t)p0;
            pbuf[wave][m * 32 + 16 + l16] = (bf16_t)p1;
        }

        // ---- reload P as A-fragment (16x32) via LDS ----
        const v8bf plo = *(const v8bf*)(&pbuf[wave][l16 * 32 + half * 16]);
        const v8bf phi = *(const v8bf*)(&pbuf[wave][l16 * 32 + half * 16 + 8]);
        const v16bf pa = __builtin_shufflevector(plo, phi,
            0,1,2,3,4,5,6,7,8,9,10,11,12,13,14,15);

        // ---- burst: O += P * V ----
        #pragma unroll
        for (int vt = 0; vt < 8; ++vt)
            o_acc[vt] = __builtin_amdgcn_wmma_f32_16x16x32_bf16(
                false, pa, false, vf[vt], (short)0, o_acc[vt], false, false);
    }

    // ---- normalize and write output [B, T, 128] f32 ----
    #pragma unroll
    for (int r = 0; r < 8; ++r) {
        const float inv_l = 1.0f / l_st[r];
        const int m = r + half * 8;
        #pragma unroll
        for (int vt = 0; vt < 8; ++vt)
            out[((size_t)b * T + q0 + m) * D + vt * 16 + l16] =
                o_acc[vt][r] * inv_l;
    }
}

// ---------------------------------------------------------------------------
extern "C" void kernel_launch(void* const* d_in, const int* in_sizes, int n_in,
                              void* d_out, int out_size, void* d_ws, size_t ws_size,
                              hipStream_t stream)
{
    const float* x  = (const float*)d_in[0];
    const float* Wq = (const float*)d_in[1];
    const float* Wk = (const float*)d_in[2];
    const float* Wv = (const float*)d_in[3];

    // Workspace layout (bf16, all 32B-aligned):
    //   xb (32 MB) | Wqb | Wkb | Wvb (256 KB each) | Qb (4 MB) | Kb (4 MB) | Vt (4 MB)
    bf16_t* xb  = (bf16_t*)d_ws;
    bf16_t* Wqb = xb  + (size_t)NB * T * C;
    bf16_t* Wkb = Wqb + (size_t)D * C;
    bf16_t* Wvb = Wkb + (size_t)D * C;
    bf16_t* Qb  = Wvb + (size_t)D * C;
    bf16_t* Kb  = Qb  + (size_t)NB * T * D;
    bf16_t* Vt  = Kb  + (size_t)NB * T * D;

    const int nx = NB * T * C;          // 16,777,216
    const int nw = D * C;               // 131,072
    cvt_f32_to_bf16_kernel<<<nx / (256 * 8), 256, 0, stream>>>(x,  xb,  nx);
    cvt_f32_to_bf16_kernel<<<nw / (256 * 8), 256, 0, stream>>>(Wq, Wqb, nw);
    cvt_f32_to_bf16_kernel<<<nw / (256 * 8), 256, 0, stream>>>(Wk, Wkb, nw);
    cvt_f32_to_bf16_kernel<<<nw / (256 * 8), 256, 0, stream>>>(Wv, Wvb, nw);

    // 3 matrices x 1024 row-tiles = 3072 wave tasks, 8 waves/block.
    proj_qkv_kernel<<<384, 256, 0, stream>>>(xb, Wqb, Wkb, Wvb, Qb, Kb, Vt);

    // 1024 query tiles, 8 waves/block.
    flash_attn_kernel<<<128, 256, 0, stream>>>(Qb, Kb, Vt, (float*)d_out);
}
